// HROMAttention_80126909874290
// MI455X (gfx1250) — matrix operand order repaired and checked by reading, not compiled, not run on hardware
//
#include <hip/hip_runtime.h>
#include <hip/hip_bf16.h>

// ---------------------------------------------------------------------------
// CDNA5 (gfx1250) fused attention: bf16 WMMA (v_wmma_f32_16x16x32_bf16),
// fp32 accumulation, flash-style online softmax.
// Attention uses 4-wave workgroups with K/V double-buffered in LDS via the
// gfx1250 async global->LDS path (ASYNCcnt) when available.
// ---------------------------------------------------------------------------

typedef __bf16 bf16;
typedef __attribute__((ext_vector_type(16))) __bf16 bf16x16;
typedef __attribute__((ext_vector_type(8)))  __bf16 bf16x8;
typedef __attribute__((ext_vector_type(8)))  float  f32x8;
typedef __attribute__((ext_vector_type(4)))  int    i32x4;

typedef __attribute__((address_space(1))) i32x4 g_i32x4;   // global
typedef __attribute__((address_space(3))) i32x4 l_i32x4;   // LDS

#define DIM      512
#define NHEADS   8
#define HDIM     64
#define SEQ      2048
#define BATCH    4

__device__ __forceinline__ f32x8 wmma_bf16(bf16x16 a, bf16x16 b, f32x8 c) {
  // 8 args: (neg_a, A, neg_b, B, c_mod, C, reuse_a, reuse_b)
  return __builtin_amdgcn_wmma_f32_16x16x32_bf16(false, a, false, b,
                                                 (short)0, c, false, false);
}

// A-matrix 16x32 bf16 fragment, row-major source, row = lane&15.
// lanes 0-15 hold K=[0..7],[16..23]; lanes 16-31 hold K=[8..15],[24..31].
__device__ __forceinline__ bf16x16 load_frag_a(const bf16* base, int row0,
                                               int ld, int k0) {
  int lane = threadIdx.x & 31;
  int r = lane & 15, half = lane >> 4;
  const bf16* p = base + (size_t)(row0 + r) * ld + k0 + half * 8;
  union { bf16x16 f; bf16x8 h[2]; } u;
  u.h[0] = *(const bf16x8*)(p);
  u.h[1] = *(const bf16x8*)(p + 16);
  return u.f;
}

// B-matrix 32x16 bf16 fragment: column n = lane&15; lanes 0-15 hold
// K=[0..7],[16..23]; lanes 16-31 hold K=[8..15],[24..31]; column n's K-run at base + (n0+n)*ld + k0.
__device__ __forceinline__ bf16x16 load_frag_b(const bf16* base, int n0,
                                               int ld, int k0) {
  int lane = threadIdx.x & 31;
  int n = lane & 15, half = lane >> 4;
  const bf16* p = base + (size_t)(n0 + n) * ld + k0 + half * 8;
  union { bf16x16 f; bf16x8 h[2]; } u;
  u.h[0] = *(const bf16x8*)(p);
  u.h[1] = *(const bf16x8*)(p + 16);
  return u.f;
}

// 16-byte global -> LDS copy. Uses gfx1250 async DMA path when the toolchain
// exposes it (tracked by ASYNCcnt); falls back to a synchronous copy.
__device__ __forceinline__ void copy16_g2l(bf16* lds, const bf16* g) {
#if __has_builtin(__builtin_amdgcn_global_load_async_to_lds_b128)
  __builtin_amdgcn_global_load_async_to_lds_b128(
      (g_i32x4*)(unsigned long long)g,
      (l_i32x4*)(unsigned int)(unsigned long long)lds, 0, 0);
#else
  *(bf16x8*)lds = *(const bf16x8*)g;
#endif
}

template <int N>
__device__ __forceinline__ void wait_async() {
#if __has_builtin(__builtin_amdgcn_global_load_async_to_lds_b128)
#if __has_builtin(__builtin_amdgcn_s_wait_asynccnt)
  __builtin_amdgcn_s_wait_asynccnt(N);
#else
  asm volatile("s_wait_asynccnt %0" ::"n"(N) : "memory");
#endif
#endif
}

// ---------------------------------------------------------------------------
__global__ void cvt_f32_bf16_kernel(const float* __restrict__ in,
                                    bf16* __restrict__ out, int n) {
  int i = blockIdx.x * blockDim.x + threadIdx.x;
  if (i < n) out[i] = (bf16)in[i];
}

// ---------------------------------------------------------------------------
// Fused QKV projection + bias + RoPE.
// grid = (24 column-groups of 64, 512 row-tiles of 16), block = 32 (1 wave).
// Column group g covers out-features [g*64, g*64+64) == one (qkv, head) pair.
// Q,K -> bf16 [B,H,T,64] (roped). V -> bf16 [B,H,64,T] (transposed).
// ---------------------------------------------------------------------------
__global__ void qkv_rope_kernel(const bf16* __restrict__ Xb,
                                const bf16* __restrict__ Wb,
                                const float* __restrict__ bias,
                                bf16* __restrict__ Qb,
                                bf16* __restrict__ Kb,
                                bf16* __restrict__ Vt) {
  const int group = blockIdx.x;       // 0..23
  const int row0  = blockIdx.y * 16;  // global row in [0, 8192)
  const int ofb   = group * 64;
  const int lane  = threadIdx.x & 31;
  const int nn    = lane & 15, half = lane >> 4;

  f32x8 acc[4] = {};
  for (int k0 = 0; k0 < DIM; k0 += 32) {
    bf16x16 a = load_frag_a(Xb, row0, DIM, k0);
#pragma unroll
    for (int t = 0; t < 4; ++t) {
      bf16x16 b = load_frag_b(Wb, ofb + t * 16, DIM, k0);
      acc[t] = wmma_bf16(a, b, acc[t]);
    }
  }
#pragma unroll
  for (int t = 0; t < 4; ++t) {
    float bv = bias[ofb + t * 16 + nn];
#pragma unroll
    for (int i = 0; i < 8; ++i) acc[t][i] += bv;
  }

  const int qkv   = group >> 3;       // 0=q 1=k 2=v
  const int h     = group & 7;
  const int b     = row0 >> 11;
  const int tbase = row0 & (SEQ - 1);

  if (qkv < 2) {
    // RoPE: pair (d, d+32) == (tile jt, tile jt+2), angle = t * 10000^(-j/32)
    const float kfreq = -0.28782313662425575f; // -ln(10000)/32
#pragma unroll
    for (int i = 0; i < 8; ++i) {
      float tpos = (float)(tbase + i + 8 * half);
#pragma unroll
      for (int jt = 0; jt < 2; ++jt) {
        int j = jt * 16 + nn;                // d % 32
        float ang = tpos * __expf(kfreq * (float)j);
        float c = __cosf(ang), s = __sinf(ang);
        float x1 = acc[jt][i], x2 = acc[jt + 2][i];
        acc[jt][i]     = x1 * c - x2 * s;
        acc[jt + 2][i] = x2 * c + x1 * s;
      }
    }
    bf16* dst = (qkv == 0 ? Qb : Kb) + ((size_t)(b * NHEADS + h) * SEQ) * HDIM;
#pragma unroll
    for (int t = 0; t < 4; ++t)
#pragma unroll
      for (int i = 0; i < 8; ++i) {
        int trow = tbase + i + 8 * half;
        dst[(size_t)trow * HDIM + t * 16 + nn] = (bf16)acc[t][i];
      }
  } else {
    // V transposed [B,H,64,T]; per (lane,tile) the 8 rows are contiguous in t.
    bf16* dst = Vt + ((size_t)(b * NHEADS + h) * HDIM) * SEQ;
#pragma unroll
    for (int t = 0; t < 4; ++t) {
      int d = t * 16 + nn;
      bf16x8 pk;
#pragma unroll
      for (int i = 0; i < 8; ++i) pk[i] = (bf16)acc[t][i];
      *(bf16x8*)(dst + (size_t)d * SEQ + tbase + 8 * half) = pk;
    }
  }
}

// ---------------------------------------------------------------------------
// Stage one 32-key chunk of K (32x64, row-major [t][d]) and V (64x32,
// [d][t]) into LDS. 128 threads; each wave issues 4 async-load instructions.
// ---------------------------------------------------------------------------
__device__ __forceinline__ void stage_kv(const bf16* Kp, const bf16* Vp,
                                         bf16* Kl, bf16* Vl, int kc, int tid) {
#pragma unroll
  for (int rep = 0; rep < 2; ++rep) {            // K: 4KB contiguous
    int u = tid + rep * 128;
    copy16_g2l(Kl + u * 8, Kp + (size_t)kc * HDIM + u * 8);
  }
#pragma unroll
  for (int rep = 0; rep < 2; ++rep) {            // V: 64 rows x 64B
    int u = tid + rep * 128;
    int d = u >> 2, part = u & 3;
    copy16_g2l(Vl + d * 32 + part * 8, Vp + (size_t)d * SEQ + kc + part * 8);
  }
}

// ---------------------------------------------------------------------------
// Flash attention. grid = (B*H = 32, T/64 = 32), block = 128 (4 waves).
// Wave w owns query rows [q0 + 16w, q0 + 16w + 16). K/V chunks of 32 keys are
// double-buffered in LDS (async DMA) and shared by all 4 waves.
// ---------------------------------------------------------------------------
__global__ void flash_attn_kernel(const bf16* __restrict__ Qb,
                                  const bf16* __restrict__ Kb,
                                  const bf16* __restrict__ Vt,
                                  const float* __restrict__ mask,
                                  bf16* __restrict__ Ob) {
  __shared__ __align__(16) bf16 Kl[2][32 * HDIM];   // 2 x 4KB
  __shared__ __align__(16) bf16 Vl[2][HDIM * 32];   // 2 x 4KB
  __shared__ __align__(16) bf16 Pl[4][16 * 32];     // per-wave P tile

  const int bh   = blockIdx.x;                 // 0..31
  const int b    = bh >> 3, h = bh & 7;
  const int tid  = threadIdx.x;                // 0..127
  const int w    = tid >> 5;                   // wave id 0..3
  const int q0   = blockIdx.y * 64 + w * 16;
  const int lane = tid & 31;
  const int nn   = lane & 15, half = lane >> 4;

  const bf16*  Qp = Qb + (size_t)bh * SEQ * HDIM;
  const bf16*  Kp = Kb + (size_t)bh * SEQ * HDIM;
  const bf16*  Vp = Vt + (size_t)bh * HDIM * SEQ;
  const float* mp = mask + (size_t)b * SEQ * SEQ;

  // Q fragments, reused for the whole key loop
  bf16x16 qa0 = load_frag_a(Qp, q0, HDIM, 0);
  bf16x16 qa1 = load_frag_a(Qp, q0, HDIM, 32);

  float m_i[8], l_i[8];
  f32x8 acc[4] = {};
#pragma unroll
  for (int i = 0; i < 8; ++i) { m_i[i] = -1e30f; l_i[i] = 0.0f; }

  const float scale = 0.125f;  // 1/sqrt(64)
  const int   NC    = SEQ / 32;

  stage_kv(Kp, Vp, Kl[0], Vl[0], 0, tid);

  for (int c = 0; c < NC; ++c) {
    const int cur = c & 1, nxt = cur ^ 1;
    const int kc  = c * 32;

    if (c + 1 < NC) {
      stage_kv(Kp, Vp, Kl[nxt], Vl[nxt], kc + 32, tid);
      wait_async<4>();        // chunk c landed; chunk c+1 may stay in flight
    } else {
      wait_async<0>();
    }
    __syncthreads();          // all waves' quarters of buf[cur] are visible

    const bf16* Klp = Kl[cur];
    const bf16* Vlp = Vl[cur];

    f32x8 s0 = {}, s1 = {};
    s0 = wmma_bf16(qa0, load_frag_b(Klp, 0,  HDIM, 0),  s0);
    s0 = wmma_bf16(qa1, load_frag_b(Klp, 0,  HDIM, 32), s0);
    s1 = wmma_bf16(qa0, load_frag_b(Klp, 16, HDIM, 0),  s1);
    s1 = wmma_bf16(qa1, load_frag_b(Klp, 16, HDIM, 32), s1);

    // scale + mask  (D-layout: row = i + 8*half, col = nn)
#pragma unroll
    for (int i = 0; i < 8; ++i) {
      size_t qrow = (size_t)(q0 + i + 8 * half);
      s0[i] = s0[i] * scale + mp[qrow * SEQ + kc + nn];
      s1[i] = s1[i] * scale + mp[qrow * SEQ + kc + 16 + nn];
    }

    // online softmax: per-row stats via 16-lane-half butterflies
    float alpha[8];
#pragma unroll
    for (int i = 0; i < 8; ++i) {
      float mx = fmaxf(s0[i], s1[i]);
#pragma unroll
      for (int o = 8; o >= 1; o >>= 1) mx = fmaxf(mx, __shfl_xor(mx, o, 32));
      float mnew = fmaxf(m_i[i], mx);
      float a = __expf(m_i[i] - mnew);
      m_i[i] = mnew;
      float p0 = __expf(s0[i] - mnew);
      float p1 = __expf(s1[i] - mnew);
      s0[i] = p0; s1[i] = p1;
      float rs = p0 + p1;
#pragma unroll
      for (int o = 8; o >= 1; o >>= 1) rs += __shfl_xor(rs, o, 32);
      l_i[i] = l_i[i] * a + rs;
      alpha[i] = a;
    }
#pragma unroll
    for (int t = 0; t < 4; ++t)
#pragma unroll
      for (int i = 0; i < 8; ++i) acc[t][i] *= alpha[i];

    // P (16x32 bf16) through wave-private LDS: D-layout -> A-layout
    bf16* Pw = Pl[w];
#pragma unroll
    for (int i = 0; i < 8; ++i) {
      int r = i + 8 * half;
      Pw[r * 32 + nn]      = (bf16)s0[i];
      Pw[r * 32 + 16 + nn] = (bf16)s1[i];
    }
    bf16x16 pa = load_frag_a(Pw, 0, 32, 0);

    // O += P * V   (V in LDS as [d][t]: column n = head-dim, contiguous keys)
#pragma unroll
    for (int t = 0; t < 4; ++t) {
      bf16x16 vb = load_frag_b(Vlp, t * 16, 32, 0);
      acc[t] = wmma_bf16(pa, vb, acc[t]);
    }

    __syncthreads();          // buf[cur] free before iteration c+1 re-stages it
  }

  // normalize and scatter to [B,T,512] bf16 at (b, t, h*64 + d)
#pragma unroll
  for (int t = 0; t < 4; ++t)
#pragma unroll
    for (int i = 0; i < 8; ++i) {
      int trow = q0 + i + 8 * half;
      float o = acc[t][i] / l_i[i];
      Ob[((size_t)b * SEQ + trow) * DIM + h * HDIM + t * 16 + nn] = (bf16)o;
    }
}

// ---------------------------------------------------------------------------
// Output projection: out = AttnOut(8192x512) @ w_proj^T + b_proj  (fp32 out)
// grid = (8 column-groups of 64, 512 row-tiles), block = 32.
// ---------------------------------------------------------------------------
__global__ void proj_kernel(const bf16* __restrict__ Ob,
                            const bf16* __restrict__ Wb,
                            const float* __restrict__ bias,
                            float* __restrict__ out) {
  const int group = blockIdx.x;
  const int row0  = blockIdx.y * 16;
  const int ofb   = group * 64;
  const int lane  = threadIdx.x & 31;
  const int nn = lane & 15, half = lane >> 4;

  f32x8 acc[4] = {};
  for (int k0 = 0; k0 < DIM; k0 += 32) {
    bf16x16 a = load_frag_a(Ob, row0, DIM, k0);
#pragma unroll
    for (int t = 0; t < 4; ++t) {
      bf16x16 b = load_frag_b(Wb, ofb + t * 16, DIM, k0);
      acc[t] = wmma_bf16(a, b, acc[t]);
    }
  }
#pragma unroll
  for (int t = 0; t < 4; ++t) {
    float bv = bias[ofb + t * 16 + nn];
#pragma unroll
    for (int i = 0; i < 8; ++i) {
      int row = row0 + i + 8 * half;
      out[(size_t)row * DIM + ofb + t * 16 + nn] = acc[t][i] + bv;
    }
  }
}

// ---------------------------------------------------------------------------
extern "C" void kernel_launch(void* const* d_in, const int* in_sizes, int n_in,
                              void* d_out, int out_size, void* d_ws,
                              size_t ws_size, hipStream_t stream) {
  const float* x      = (const float*)d_in[0];
  const float* mask   = (const float*)d_in[1];
  const float* w_qkv  = (const float*)d_in[2];
  const float* b_qkv  = (const float*)d_in[3];
  const float* w_proj = (const float*)d_in[4];
  const float* b_proj = (const float*)d_in[5];
  float* out = (float*)d_out;

  char* ws = (char*)d_ws;
  const int nX  = BATCH * SEQ * DIM;        // 4194304
  const int nWq = 3 * DIM * DIM;            // 786432
  const int nWp = DIM * DIM;                // 262144
  bf16* Xb     = (bf16*)(ws);                         //  8.0 MB
  bf16* Wqkvb  = (bf16*)(ws +  8388608);              //  1.5 MB
  bf16* Wprojb = (bf16*)(ws +  9961472);              //  0.5 MB
  bf16* Qb     = (bf16*)(ws + 10485760);              //  8.0 MB
  bf16* Kb     = (bf16*)(ws + 18874368);              //  8.0 MB
  bf16* Vt     = (bf16*)(ws + 27262976);              //  8.0 MB
  bf16* Ob     = (bf16*)(ws + 35651584);              //  8.0 MB

  cvt_f32_bf16_kernel<<<(nX  + 255) / 256, 256, 0, stream>>>(x, Xb, nX);
  cvt_f32_bf16_kernel<<<(nWq + 255) / 256, 256, 0, stream>>>(w_qkv, Wqkvb, nWq);
  cvt_f32_bf16_kernel<<<(nWp + 255) / 256, 256, 0, stream>>>(w_proj, Wprojb, nWp);

  qkv_rope_kernel<<<dim3(24, (BATCH * SEQ) / 16), 32, 0, stream>>>(
      Xb, Wqkvb, b_qkv, Qb, Kb, Vt);

  flash_attn_kernel<<<dim3(BATCH * NHEADS, SEQ / 64), 128, 0, stream>>>(
      Qb, Kb, Vt, mask, Ob);

  proj_kernel<<<dim3(DIM / 64, (BATCH * SEQ) / 16), 32, 0, stream>>>(
      Ob, Wprojb, b_proj, out);
}
